// MolecularEncoder_25520695673003
// MI455X (gfx1250) — matrix operand order repaired and checked by reading, not compiled, not run on hardware
//
#include <hip/hip_runtime.h>
#include <hip/hip_bf16.h>
#include <hip/hip_fp16.h>

// ---------------------------------------------------------------------------
// GIN molecular encoder for MI455X (gfx1250, wave32).
// GEMMs: v_wmma_f32_16x16x32_f16, one wave per 16x128 output stripe,
// A fragments loaded once per k-step (8x reuse), B staged in LDS (ds path).
// Scatter/pool/BN-stats use hardware f32 global atomics.
// ---------------------------------------------------------------------------

typedef __attribute__((ext_vector_type(16))) _Float16 v16h;
typedef __attribute__((ext_vector_type(8)))  float    v8f;

union FragH { uint4 q[2]; v16h v; };

#define HDIM  128
#define MAXV  119
#define KCHUNK 128
#define BPAD   8            // halfs of row padding in LDS (bank spread)
#define BSTRIDE (KCHUNK + BPAD)

__device__ __forceinline__ void atomAddF(float* p, float v) {
  __hip_atomic_fetch_add(p, v, __ATOMIC_RELAXED, __HIP_MEMORY_SCOPE_AGENT);
}

// --- weight convert: src f32 [K x Nc] row-major -> dst f16 [Nc x K] ---------
__global__ __launch_bounds__(256)
void k_transpose_f16(const float* __restrict__ src, _Float16* __restrict__ dst,
                     int K, int Nc) {
  long i = (long)blockIdx.x * blockDim.x + threadIdx.x;
  if (i >= (long)K * Nc) return;
  int k = (int)(i / Nc), n = (int)(i % Nc);
  dst[(long)n * K + k] = (_Float16)src[i];
}

// --- atom encoder: h[n] = sum_s emb[s][x[n][s]]  (one wave per node) --------
__global__ __launch_bounds__(256)
void k_atom_encode(const int* __restrict__ x, const float* __restrict__ emb,
                   float* __restrict__ h, int N) {
  int wave = (int)((blockIdx.x * (long)blockDim.x + threadIdx.x) >> 5);
  int lane = threadIdx.x & 31;
  if (wave >= N) return;
  float4 acc = make_float4(0.f, 0.f, 0.f, 0.f);
  for (int s = 0; s < 9; ++s) {
    int idx = x[wave * 9 + s];
    const float4 e = *(const float4*)(emb + ((long)s * MAXV + idx) * HDIM + lane * 4);
    acc.x += e.x; acc.y += e.y; acc.z += e.z; acc.w += e.w;
  }
  *(float4*)(h + (long)wave * HDIM + lane * 4) = acc;
}

// --- scatter-add over edges: agg[dst] += h[src]  (one wave per edge) --------
__global__ __launch_bounds__(256)
void k_scatter(const int* __restrict__ ei, const float* __restrict__ h,
               float* __restrict__ agg, int E) {
  int wave = (int)((blockIdx.x * (long)blockDim.x + threadIdx.x) >> 5);
  int lane = threadIdx.x & 31;
  if (wave >= E) return;
  int src = ei[wave];
  int dst = ei[(long)E + wave];
  const float4 v = *(const float4*)(h + (long)src * HDIM + lane * 4);
  float* p = agg + (long)dst * HDIM + lane * 4;
  atomAddF(p + 0, v.x); atomAddF(p + 1, v.y);
  atomAddF(p + 2, v.z); atomAddF(p + 3, v.w);
}

// --- z = (1+eps[l])*h + agg, converted to f16 -------------------------------
__global__ __launch_bounds__(256)
void k_prep_z(const float* __restrict__ h, const float* __restrict__ agg,
              const float* __restrict__ eps, int l, _Float16* __restrict__ zh,
              long n) {
  long i = (long)blockIdx.x * blockDim.x + threadIdx.x;
  if (i >= n) return;
  float e = 1.0f + eps[l];
  zh[i] = (_Float16)(e * h[i] + agg[i]);
}

// --- WMMA GEMM: C[M x Nc] = A[M x K] * Bt^T + bias --------------------------
// Block = 256 thr (8 waves). Block tile: 128 rows x 128 cols.
// Each wave computes one 16x128 stripe: 8 f32 accumulators, A fragment loaded
// once per 32-wide k-step and reused across all 8 WMMAs. B (transposed,
// [Nc x K] row-major) is staged in LDS in K-chunks of 128.
__global__ __launch_bounds__(256)
void k_gemm_wmma(const _Float16* __restrict__ A, const _Float16* __restrict__ Bt,
                 const float* __restrict__ bias, float* __restrict__ Cf,
                 _Float16* __restrict__ Ch, float* __restrict__ stats,
                 int M, int K, int Nc, int relu) {
  __shared__ _Float16 ldsB[128 * BSTRIDE];          // 34 KB of 320 KB WGP LDS
  const int tid  = threadIdx.x;
  const int lane = tid & 31;
  const int wave = tid >> 5;                        // 0..7
  const int r    = lane & 15;
  const int hi   = lane >> 4;                       // half-wave select
  const int rowBase = blockIdx.x * 128 + wave * 16;
  const int colBase = blockIdx.y * 128;
  int arow = rowBase + r; if (arow >= M) arow = M - 1;
  // A frag per ISA: lanes 0-15 hold K {0..7,16..23}, lanes 16-31 {8..15,24..31}
  const _Float16* ap = A + (long)arow * K + hi * 8;

  v8f zero = {};
  v8f acc[8];
  #pragma unroll
  for (int t = 0; t < 8; ++t) acc[t] = zero;

  const int bcRow  = tid >> 1;                      // col (0..127) to stage
  const int bcPart = (tid & 1) * 64;                // half-row in halfs

  for (int kc = 0; kc < K; kc += KCHUNK) {
    // ---- cooperative stage of B chunk into LDS ----
    {
      const _Float16* src = Bt + (long)(colBase + bcRow) * K + kc + bcPart;
      _Float16* dst = ldsB + bcRow * BSTRIDE + bcPart;
      #pragma unroll
      for (int j = 0; j < 64; j += 8)
        *(uint4*)(dst + j) = *(const uint4*)(src + j);
    }
    __syncthreads();
    #pragma unroll
    for (int k0 = 0; k0 < KCHUNK; k0 += 32) {
      FragH a;
      a.q[0] = *(const uint4*)(ap + kc + k0);       // K = k0+hi*8    .. +7
      a.q[1] = *(const uint4*)(ap + kc + k0 + 16);  // K = k0+hi*8+16 .. +23
      #pragma unroll
      for (int t = 0; t < 8; ++t) {
        // B frag per ISA: lanes 0-15 K {0..7,16..23}, lanes 16-31 {8..15,24..31}
        FragH b;
        const _Float16* bp = ldsB + (t * 16 + r) * BSTRIDE + k0 + hi * 8;
        b.q[0] = *(const uint4*)(bp);
        b.q[1] = *(const uint4*)(bp + 16);
        acc[t] = __builtin_amdgcn_wmma_f32_16x16x32_f16(
            /*neg_a=*/false, a.v, /*neg_b=*/false, b.v,
            /*c_mod=*/(short)0, acc[t], /*reuse_a=*/false, /*reuse_b=*/false);
      }
    }
    __syncthreads();
  }

  // ---- epilogue: bias, ReLU, stores, fused BN column stats ----
  #pragma unroll
  for (int t = 0; t < 8; ++t) {
    const int col = colBase + t * 16 + r;
    const float bv = bias ? bias[col] : 0.0f;
    float s = 0.f, s2 = 0.f;
    #pragma unroll
    for (int i = 0; i < 8; ++i) {
      int row = rowBase + i + (hi << 3);            // C layout: VGPR i -> M
      float v = acc[t][i] + bv;
      if (relu) v = fmaxf(v, 0.0f);
      if (row < M) {
        if (Cf) Cf[(long)row * Nc + col] = v;
        if (Ch) Ch[(long)row * Nc + col] = (_Float16)v;
        s += v; s2 += v * v;
      }
    }
    if (stats) {
      atomAddF(&stats[col], s);
      atomAddF(&stats[HDIM + col], s2);
    }
  }
}

// --- BN: turn column sums into scale/shift ----------------------------------
__global__ void k_bn_stats(const float* __restrict__ stats,
                           const float* __restrict__ gamma,
                           const float* __restrict__ beta, int l,
                           float* __restrict__ bnp, float invN) {
  int c = threadIdx.x;                // HDIM threads
  float mu  = stats[c] * invN;
  float var = stats[HDIM + c] * invN - mu * mu;
  float sc  = gamma[l * HDIM + c] * rsqrtf(var + 1e-5f);
  bnp[c] = sc;
  bnp[HDIM + c] = beta[l * HDIM + c] - mu * sc;
}

__global__ __launch_bounds__(256)
void k_bn_apply(const float* __restrict__ z2, const float* __restrict__ bnp,
                float* __restrict__ h, long n) {
  long i = (long)blockIdx.x * blockDim.x + threadIdx.x;
  if (i >= n) return;
  int c = (int)(i & (HDIM - 1));
  h[i] = fmaxf(z2[i] * bnp[c] + bnp[HDIM + c], 0.0f);
}

// --- global mean pool: atomics into per-graph sums + counts -----------------
__global__ __launch_bounds__(256)
void k_pool(const float* __restrict__ h, const int* __restrict__ batch,
            float* __restrict__ sums, float* __restrict__ cnt, int N) {
  int wave = (int)((blockIdx.x * (long)blockDim.x + threadIdx.x) >> 5);
  int lane = threadIdx.x & 31;
  if (wave >= N) return;
  int b = batch[wave];
  const float4 v = *(const float4*)(h + (long)wave * HDIM + lane * 4);
  float* p = sums + (long)b * HDIM + lane * 4;
  atomAddF(p + 0, v.x); atomAddF(p + 1, v.y);
  atomAddF(p + 2, v.z); atomAddF(p + 3, v.w);
  if (lane == 0) atomAddF(&cnt[b], 1.0f);
}

__global__ __launch_bounds__(256)
void k_gdiv(const float* __restrict__ sums, const float* __restrict__ cnt,
            _Float16* __restrict__ gh, long n) {
  long i = (long)blockIdx.x * blockDim.x + threadIdx.x;
  if (i >= n) return;
  float c = fmaxf(cnt[i >> 7], 1.0f);
  gh[i] = (_Float16)(sums[i] / c);
}

// --- L2 normalize rows of d_out in place (one wave per graph) ---------------
__global__ __launch_bounds__(256)
void k_l2norm(float* __restrict__ out, int B) {
  int wave = (int)((blockIdx.x * (long)blockDim.x + threadIdx.x) >> 5);
  int lane = threadIdx.x & 31;
  if (wave >= B) return;
  float4 v = *(float4*)(out + (long)wave * HDIM + lane * 4);
  float s = v.x * v.x + v.y * v.y + v.z * v.z + v.w * v.w;
  #pragma unroll
  for (int o = 16; o >= 1; o >>= 1) s += __shfl_xor(s, o, 32);
  float inv = 1.0f / fmaxf(sqrtf(s), 1e-12f);
  v.x *= inv; v.y *= inv; v.z *= inv; v.w *= inv;
  *(float4*)(out + (long)wave * HDIM + lane * 4) = v;
}

// ---------------------------------------------------------------------------
static inline size_t alignUp(size_t v, size_t a) { return (v + a - 1) / a * a; }

extern "C" void kernel_launch(void* const* d_in, const int* in_sizes, int n_in,
                              void* d_out, int out_size, void* d_ws, size_t ws_size,
                              hipStream_t stream) {
  (void)n_in; (void)ws_size;
  const int*   x     = (const int*)d_in[0];
  const int*   ei    = (const int*)d_in[1];
  const int*   batch = (const int*)d_in[2];
  const float* emb   = (const float*)d_in[3];
  const float* eps   = (const float*)d_in[4];
  const float* w1    = (const float*)d_in[5];
  const float* b1    = (const float*)d_in[6];
  const float* w2    = (const float*)d_in[7];
  const float* b2    = (const float*)d_in[8];
  const float* gamma = (const float*)d_in[9];
  const float* beta  = (const float*)d_in[10];
  const float* pw1   = (const float*)d_in[11];
  const float* pb1   = (const float*)d_in[12];
  const float* pw2   = (const float*)d_in[13];
  const float* pb2   = (const float*)d_in[14];

  const int N  = in_sizes[0] / 9;
  const int E  = in_sizes[1] / 2;
  const int L  = in_sizes[4];
  const int PH = in_sizes[12];          // 512
  const int Bg = out_size / HDIM;       // #graphs
  const long NH = (long)N * HDIM;

  // ---- carve workspace ----
  char*  base = (char*)d_ws;
  size_t off  = 0;
  auto carve = [&](size_t bytes) -> void* {
    void* p = base + off; off = alignUp(off + bytes, 256); return p;
  };
  float*     h    = (float*)carve(NH * 4);
  float*     agg  = (float*)carve(NH * 4);        // reused as z2
  _Float16*  zh   = (_Float16*)carve(NH * 2);
  _Float16*  y1h  = (_Float16*)carve(NH * 2);
  _Float16*  w1t  = (_Float16*)carve((size_t)L * HDIM * HDIM * 2);
  _Float16*  w2t  = (_Float16*)carve((size_t)L * HDIM * HDIM * 2);
  _Float16*  pw1t = (_Float16*)carve((size_t)HDIM * PH * 2);
  _Float16*  pw2t = (_Float16*)carve((size_t)PH * HDIM * 2);
  float*     stats = (float*)carve(2 * HDIM * 4);
  float*     bnp   = (float*)carve(2 * HDIM * 4);
  float*     sums  = (float*)carve((size_t)Bg * (HDIM + 1) * 4); // sums + cnt
  float*     cnt   = sums + (size_t)Bg * HDIM;
  _Float16*  gh    = (_Float16*)carve((size_t)Bg * HDIM * 2);
  _Float16*  p1h   = (_Float16*)carve((size_t)Bg * PH * 2);
  float*     gout  = (float*)d_out;

  const int ew = 256;                   // elementwise block
  const int wgrid_N = (N + 7) / 8;      // one wave per node, 8 waves/block

  // ---- weight conversion + transpose (f32 -> f16, [K x N] -> [N x K]) ----
  {
    int hh = HDIM * HDIM;
    for (int l = 0; l < L; ++l) {
      k_transpose_f16<<<(hh + ew - 1) / ew, ew, 0, stream>>>(w1 + (long)l * hh, w1t + (long)l * hh, HDIM, HDIM);
      k_transpose_f16<<<(hh + ew - 1) / ew, ew, 0, stream>>>(w2 + (long)l * hh, w2t + (long)l * hh, HDIM, HDIM);
    }
    int hp = HDIM * PH;
    k_transpose_f16<<<(hp + ew - 1) / ew, ew, 0, stream>>>(pw1, pw1t, HDIM, PH);
    k_transpose_f16<<<(hp + ew - 1) / ew, ew, 0, stream>>>(pw2, pw2t, PH, HDIM);
  }

  // ---- atom encoder ----
  k_atom_encode<<<wgrid_N, ew, 0, stream>>>(x, emb, h, N);

  // ---- GIN layers ----
  const dim3 gemmGridH((N + 127) / 128, HDIM / 128);   // y = 1
  for (int l = 0; l < L; ++l) {
    hipMemsetAsync(agg, 0, NH * 4, stream);
    k_scatter<<<(E + 7) / 8, ew, 0, stream>>>(ei, h, agg, E);
    k_prep_z<<<(int)((NH + ew - 1) / ew), ew, 0, stream>>>(h, agg, eps, l, zh, NH);
    // z @ w1 + b1, ReLU -> f16
    k_gemm_wmma<<<gemmGridH, 256, 0, stream>>>(zh, w1t + (long)l * HDIM * HDIM,
        b1 + (long)l * HDIM, nullptr, y1h, nullptr, N, HDIM, HDIM, 1);
    hipMemsetAsync(stats, 0, 2 * HDIM * 4, stream);
    // y1 @ w2 + b2 -> f32 (z2 aliases agg) with fused BN column stats
    k_gemm_wmma<<<gemmGridH, 256, 0, stream>>>(y1h, w2t + (long)l * HDIM * HDIM,
        b2 + (long)l * HDIM, agg, nullptr, stats, N, HDIM, HDIM, 0);
    k_bn_stats<<<1, HDIM, 0, stream>>>(stats, gamma, beta, l, bnp, 1.0f / (float)N);
    k_bn_apply<<<(int)((NH + ew - 1) / ew), ew, 0, stream>>>(agg, bnp, h, NH);
  }

  // ---- global mean pool ----
  hipMemsetAsync(sums, 0, (size_t)Bg * (HDIM + 1) * 4, stream);
  k_pool<<<wgrid_N, ew, 0, stream>>>(h, batch, sums, cnt, N);
  long BH = (long)Bg * HDIM;
  k_gdiv<<<(int)((BH + ew - 1) / ew), ew, 0, stream>>>(sums, cnt, gh, BH);

  // ---- projection head ----
  k_gemm_wmma<<<dim3((Bg + 127) / 128, PH / 128), 256, 0, stream>>>(
      gh, pw1t, pb1, nullptr, p1h, nullptr, Bg, HDIM, PH, 1);
  k_gemm_wmma<<<dim3((Bg + 127) / 128, HDIM / 128), 256, 0, stream>>>(
      p1h, pw2t, pb2, gout, nullptr, nullptr, Bg, PH, HDIM, 0);

  // ---- L2 normalize in place ----
  k_l2norm<<<(Bg + 7) / 8, ew, 0, stream>>>(gout, Bg);
}
